// MambaBlock_84095459656132
// MI455X (gfx1250) — compile-verified, matrix operand order checked by reading, not run on hardware
//
#include <hip/hip_runtime.h>

// ---------------------------------------------------------------------------
// Mamba block for MI455X (gfx1250, wave32, WMMA, async global->LDS staging).
// ---------------------------------------------------------------------------

#define Bsz   2
#define Lseq  4096
#define Dmod  2048
#define Hhd   32
#define DSd   64
#define Kcv   4
#define Mrow  (Bsz * Lseq)          // 8192
#define N1    (3 * Hhd * DSd)       // 6144

#define PADK  40                    // padded row stride (halfwords) for LDS slabs

typedef __attribute__((ext_vector_type(16))) __bf16 v16bf;
typedef __attribute__((ext_vector_type(8)))  float  v8f;
typedef __attribute__((ext_vector_type(4)))  int    v4i;

// Address-space-qualified pointer types for the async copy builtin:
// param1 = v4i in AS1 (global, prints as __device__), param2 = v4i in AS3 (LDS).
typedef v4i __attribute__((address_space(1)))* gv4p;
typedef v4i __attribute__((address_space(3)))* lv4p;

// float -> bf16 bits, round-to-nearest-even
__device__ __forceinline__ unsigned short f2bf(float f) {
    unsigned int u = __float_as_uint(f);
    unsigned int r = (u + 0x7FFFu + ((u >> 16) & 1u)) >> 16;
    return (unsigned short)r;
}

__device__ __forceinline__ float sigmoidf(float v) {
    return 1.0f / (1.0f + expf(-v));
}

// --- gfx1250 async global->LDS copy (ASYNCcnt) with portable fallback ------
#if __has_builtin(__builtin_amdgcn_global_load_async_to_lds_b128)
#define HAVE_ASYNC_LDS 1
#else
#define HAVE_ASYNC_LDS 0
#endif

__device__ __forceinline__ void cp16_async(unsigned short* l, const unsigned short* g) {
#if HAVE_ASYNC_LDS
    __builtin_amdgcn_global_load_async_to_lds_b128((gv4p)g, (lv4p)l, 0, 0);
#else
    *(uint4*)l = *(const uint4*)g;
#endif
}

__device__ __forceinline__ void wait_async() {
#if HAVE_ASYNC_LDS
#if __has_builtin(__builtin_amdgcn_s_wait_asynccnt)
    __builtin_amdgcn_s_wait_asynccnt(0);
#else
    asm volatile("s_wait_asynccnt 0" ::: "memory");
#endif
#endif
}

// ---------------------------------------------------------------------------
// 1) Transpose fp32 (Kd x Nd, row-major) -> bf16 (Nd x Kd, row-major)
// ---------------------------------------------------------------------------
__global__ __launch_bounds__(256)
void transpose_to_bf16(const float* __restrict__ src,
                       unsigned short* __restrict__ dst,
                       int Kd, int Nd) {
    __shared__ float tile[32][33];
    int n0 = blockIdx.x * 32;
    int k0 = blockIdx.y * 32;
    int tx = threadIdx.x;   // 0..31
    int ty = threadIdx.y;   // 0..7
    for (int i = ty; i < 32; i += 8)
        tile[i][tx] = src[(size_t)(k0 + i) * Nd + (n0 + tx)];
    __syncthreads();
    for (int i = ty; i < 32; i += 8)
        dst[(size_t)(n0 + i) * Kd + (k0 + tx)] = f2bf(tile[tx][i]);
}

// ---------------------------------------------------------------------------
// 2) RMSNorm: one block (256 thr) per row of D=2048
// ---------------------------------------------------------------------------
__global__ __launch_bounds__(256)
void rmsnorm_kernel(const float* __restrict__ x,
                    const float* __restrict__ w,
                    float* __restrict__ xn,
                    unsigned short* __restrict__ xnb) {
    int row = blockIdx.x;
    const float* xr = x + (size_t)row * Dmod;
    float s = 0.0f;
    for (int d = threadIdx.x; d < Dmod; d += 256) {
        float v = xr[d];
        s += v * v;
    }
    for (int off = 16; off > 0; off >>= 1)
        s += __shfl_down(s, off, 32);
    __shared__ float red[8];
    int wid = threadIdx.x >> 5, lane = threadIdx.x & 31;
    if (lane == 0) red[wid] = s;
    __syncthreads();
    if (threadIdx.x == 0) {
        float tsum = 0.0f;
        for (int i = 0; i < 8; ++i) tsum += red[i];
        red[0] = rsqrtf(tsum / (float)Dmod + 1e-6f);
    }
    __syncthreads();
    float scale = red[0];
    for (int d = threadIdx.x; d < Dmod; d += 256) {
        float v = xr[d] * scale * w[d];
        xn [(size_t)row * Dmod + d] = v;
        xnb[(size_t)row * Dmod + d] = f2bf(v);
    }
}

// ---------------------------------------------------------------------------
// 3) Depthwise causal conv (K=4) + SiLU -> bf16
// ---------------------------------------------------------------------------
__global__ __launch_bounds__(256)
void conv_silu_kernel(const float* __restrict__ xn,
                      const float* __restrict__ conv_w,
                      const float* __restrict__ conv_b,
                      unsigned short* __restrict__ xcb) {
    size_t idx = (size_t)blockIdx.x * 256 + threadIdx.x;
    if (idx >= (size_t)Bsz * Lseq * Dmod) return;
    int d = (int)(idx % Dmod);
    int l = (int)((idx / Dmod) % Lseq);
    int b = (int)(idx / ((size_t)Dmod * Lseq));
    float acc = conv_b[d];
    #pragma unroll
    for (int k = 0; k < Kcv; ++k) {
        int ll = l + k - (Kcv - 1);
        if (ll >= 0)
            acc += xn[((size_t)b * Lseq + ll) * Dmod + d] * conv_w[d * Kcv + k];
    }
    acc = acc * sigmoidf(acc);   // SiLU
    xcb[idx] = f2bf(acc);
}

// ---------------------------------------------------------------------------
// 4) WMMA bf16 GEMM with double-buffered async global->LDS staging.
//    out = epilogue(A[M x K] @ Wt[N x K]^T + bias)
//    flags bit0 = sigmoid, bit1 = add residual (same shape as out)
//    Block = 256 thr (8 waves, 2x4), block tile 128x256, wave tile 64x64,
//    K-step 32. LDS slabs padded to PADK=40 halfwords/row (bank-conflict-free
//    ds_load_b128: bank-group = 5*lane mod 16 is a permutation of lanes 0..15).
// ---------------------------------------------------------------------------
union FragU {
    uint4 q[2];
    v16bf v;
};

__global__ __launch_bounds__(256)
void gemm_bf16_wmma(const unsigned short* __restrict__ A,
                    const unsigned short* __restrict__ Wt,
                    const float* __restrict__ bias,
                    const float* __restrict__ residual,
                    float* __restrict__ out,
                    int M, int N, int Kd, int flags) {
    __shared__ __align__(16) unsigned short ldsA[2][128 * PADK];
    __shared__ __align__(16) unsigned short ldsB[2][256 * PADK];

    const int t    = threadIdx.x;
    const int wid  = t >> 5;
    const int lane = t & 31;
    const int wm   = wid >> 2;          // 0..1   (64 rows each)
    const int wn   = wid & 3;           // 0..3   (64 cols each)
    const int rowB = blockIdx.y * 128;
    const int colB = blockIdx.x * 256;
    const int half = lane >> 4;         // 0 or 1
    const int sub  = lane & 15;

    // Stage one K-slab (BK=32 halfwords = 4 chunks of 16B per row) into LDS.
    auto issueA = [&](int buf, int k0) {
        #pragma unroll
        for (int i = 0; i < 2; ++i) {                 // 128 rows * 4 chunks / 256 thr
            int c = t + 256 * i;
            int r = c >> 2, cc = c & 3;
            cp16_async(&ldsA[buf][r * PADK + cc * 8],
                       A + (size_t)(rowB + r) * Kd + k0 + cc * 8);
        }
    };
    auto issueB = [&](int buf, int k0) {
        #pragma unroll
        for (int i = 0; i < 4; ++i) {                 // 256 rows * 4 chunks / 256 thr
            int c = t + 256 * i;
            int r = c >> 2, cc = c & 3;
            cp16_async(&ldsB[buf][r * PADK + cc * 8],
                       Wt + (size_t)(colB + r) * Kd + k0 + cc * 8);
        }
    };

    v8f acc[4][4] = {};

    issueA(0, 0);
    issueB(0, 0);
    int cur = 0;

    for (int k0 = 0; k0 < Kd; k0 += 32) {
        wait_async();          // my copies for buf `cur` are complete
        __syncthreads();       // everyone's copies visible; prev reads consumed

        if (k0 + 32 < Kd) {    // overlap next slab's copy with this slab's math
            issueA(cur ^ 1, k0 + 32);
            issueB(cur ^ 1, k0 + 32);
        }

        FragU af[4], bf[4];
        #pragma unroll
        for (int mt = 0; mt < 4; ++mt) {
            int base = (wm * 64 + mt * 16 + sub) * PADK + half * 8;
            af[mt].q[0] = *(const uint4*)&ldsA[cur][base];
            af[mt].q[1] = *(const uint4*)&ldsA[cur][base + 16];
        }
        #pragma unroll
        for (int nt = 0; nt < 4; ++nt) {
            int base = (wn * 64 + nt * 16 + sub) * PADK + half * 8;
            bf[nt].q[0] = *(const uint4*)&ldsB[cur][base];
            bf[nt].q[1] = *(const uint4*)&ldsB[cur][base + 16];
        }

        #pragma unroll
        for (int mt = 0; mt < 4; ++mt)
            #pragma unroll
            for (int nt = 0; nt < 4; ++nt)
                acc[mt][nt] = __builtin_amdgcn_wmma_f32_16x16x32_bf16(
                    false, af[mt].v, false, bf[nt].v,
                    (short)0, acc[mt][nt], false, false);

        cur ^= 1;
    }

    // Epilogue: C/D layout -> VGPR i holds M = i + half*8, N = sub (per tile)
    #pragma unroll
    for (int mt = 0; mt < 4; ++mt) {
        #pragma unroll
        for (int nt = 0; nt < 4; ++nt) {
            int c = colB + wn * 64 + nt * 16 + sub;
            float bv = bias[c];
            #pragma unroll
            for (int i = 0; i < 8; ++i) {
                int r = rowB + wm * 64 + mt * 16 + half * 8 + i;
                float v = acc[mt][nt][i] + bv;
                if (flags & 1) v = sigmoidf(v);
                size_t o = (size_t)r * N + c;
                if (flags & 2) v += residual[o];
                out[o] = v;
            }
        }
    }
}

// ---------------------------------------------------------------------------
// 5) SSM scan: 4096 independent lanes over (b,h,ds), sequential in L.
//    p layout: (B, L, H, 3, DS). Writes ssm (B,L,D) and h_last.
// ---------------------------------------------------------------------------
__global__ __launch_bounds__(256)
void scan_kernel(const float* __restrict__ p,
                 const float* __restrict__ state,
                 float* __restrict__ ssm,
                 float* __restrict__ h_last) {
    int t = blockIdx.x * 256 + threadIdx.x;
    if (t >= Bsz * Hhd * DSd) return;
    int b  = t / (Hhd * DSd);
    int hd = t % (Hhd * DSd);
    int h  = hd / DSd;
    int ds = hd % DSd;

    float hstate = state[t];
    const float* pb = p + (size_t)b * Lseq * Hhd * 3 * DSd + (size_t)h * 3 * DSd + ds;
    float* sb = ssm + (size_t)b * Lseq * Dmod + (size_t)h * DSd + ds;

    const size_t lstride = (size_t)Hhd * 3 * DSd;
    #pragma unroll 4
    for (int l = 0; l < Lseq; ++l) {
        const float* pr = pb + (size_t)l * lstride;
        if (l + 16 < Lseq)
            __builtin_prefetch(pr + 16 * lstride, 0, 3);  // WGP-scope stream-ahead
        float dlt = sigmoidf(pr[0]);
        float Bv  = pr[DSd];
        float Cv  = pr[2 * DSd];
        hstate = dlt * hstate + Bv;
        sb[(size_t)l * Dmod] = Cv * hstate;
    }
    h_last[t] = hstate;
}

// ---------------------------------------------------------------------------
// 6) Gated mix: mixed = gate*ssm + (1-gate)*xn  -> bf16
// ---------------------------------------------------------------------------
__global__ __launch_bounds__(256)
void mix_kernel(const float* __restrict__ gate,
                const float* __restrict__ ssm,
                const float* __restrict__ xn,
                unsigned short* __restrict__ mixedb) {
    size_t idx = (size_t)blockIdx.x * 256 + threadIdx.x;
    if (idx >= (size_t)Mrow * Dmod) return;
    float g = gate[idx];
    float v = g * ssm[idx] + (1.0f - g) * xn[idx];
    mixedb[idx] = f2bf(v);
}

// ---------------------------------------------------------------------------
// Launch
// ---------------------------------------------------------------------------
extern "C" void kernel_launch(void* const* d_in, const int* in_sizes, int n_in,
                              void* d_out, int out_size, void* d_ws, size_t ws_size,
                              hipStream_t stream) {
    const float* x      = (const float*)d_in[0];
    const float* state  = (const float*)d_in[1];
    const float* norm_w = (const float*)d_in[2];
    const float* conv_w = (const float*)d_in[3];
    const float* conv_b = (const float*)d_in[4];
    const float* pp_w   = (const float*)d_in[5];
    const float* pp_b   = (const float*)d_in[6];
    const float* gp_w   = (const float*)d_in[7];
    const float* gp_b   = (const float*)d_in[8];
    const float* op_w   = (const float*)d_in[9];
    const float* op_b   = (const float*)d_in[10];

    float* y      = (float*)d_out;                       // (B,L,D)
    float* h_last = (float*)d_out + (size_t)Mrow * Dmod; // (B,H,DS)

    // Workspace carve-up (256B aligned pieces)
    char* ws = (char*)d_ws;
    size_t off = 0;
    auto carve = [&](size_t bytes) {
        char* p = ws + off;
        off += (bytes + 255) & ~(size_t)255;
        return p;
    };
    float*          xn     = (float*)         carve((size_t)Mrow * Dmod * 4);
    unsigned short* xnb    = (unsigned short*)carve((size_t)Mrow * Dmod * 2);
    unsigned short* xcb    = (unsigned short*)carve((size_t)Mrow * Dmod * 2);
    unsigned short* wt_pp  = (unsigned short*)carve((size_t)Dmod * N1   * 2);
    unsigned short* wt_gp  = (unsigned short*)carve((size_t)Dmod * Dmod * 2);
    unsigned short* wt_op  = (unsigned short*)carve((size_t)Dmod * Dmod * 2);
    float*          p      = (float*)         carve((size_t)Mrow * N1   * 4);
    float*          gate   = (float*)         carve((size_t)Mrow * Dmod * 4);
    float*          ssm    = (float*)         carve((size_t)Mrow * Dmod * 4);
    unsigned short* mixedb = (unsigned short*)carve((size_t)Mrow * Dmod * 2);
    (void)ws_size; (void)in_sizes; (void)n_in; (void)out_size;

    // 1) Weights -> bf16 transposed
    {
        dim3 blk(32, 8);
        transpose_to_bf16<<<dim3(N1   / 32, Dmod / 32), blk, 0, stream>>>(pp_w, wt_pp, Dmod, N1);
        transpose_to_bf16<<<dim3(Dmod / 32, Dmod / 32), blk, 0, stream>>>(gp_w, wt_gp, Dmod, Dmod);
        transpose_to_bf16<<<dim3(Dmod / 32, Dmod / 32), blk, 0, stream>>>(op_w, wt_op, Dmod, Dmod);
    }

    // 2) RMSNorm
    rmsnorm_kernel<<<Mrow, 256, 0, stream>>>(x, norm_w, xn, xnb);

    // 3) Conv + SiLU
    {
        size_t total = (size_t)Mrow * Dmod;
        conv_silu_kernel<<<(unsigned)((total + 255) / 256), 256, 0, stream>>>(xn, conv_w, conv_b, xcb);
    }

    // 4a) p = xn @ pp_w + pp_b
    gemm_bf16_wmma<<<dim3(N1 / 256, Mrow / 128), 256, 0, stream>>>(
        xnb, wt_pp, pp_b, nullptr, p, Mrow, N1, Dmod, /*flags=*/0);

    // 4b) gate = sigmoid(xc @ gp_w + gp_b)
    gemm_bf16_wmma<<<dim3(Dmod / 256, Mrow / 128), 256, 0, stream>>>(
        xcb, wt_gp, gp_b, nullptr, gate, Mrow, Dmod, Dmod, /*flags=*/1);

    // 5) Scan
    scan_kernel<<<(Bsz * Hhd * DSd + 255) / 256, 256, 0, stream>>>(p, state, ssm, h_last);

    // 6) Gated mix -> bf16
    {
        size_t total = (size_t)Mrow * Dmod;
        mix_kernel<<<(unsigned)((total + 255) / 256), 256, 0, stream>>>(gate, ssm, xn, mixedb);
    }

    // 7) y = mixed @ op_w + op_b + x
    gemm_bf16_wmma<<<dim3(Dmod / 256, Mrow / 128), 256, 0, stream>>>(
        mixedb, wt_op, op_b, x, y, Mrow, Dmod, Dmod, /*flags=*/2);
}
